// HeteroAdversarialGNN_2001454760082
// MI455X (gfx1250) — matrix operand order repaired and checked by reading, not compiled, not run on hardware
//
#include <hip/hip_runtime.h>
#include <hip/hip_bf16.h>
#include <stddef.h>
#include <stdint.h>

// ---------------------------------------------------------------------------
// Types for CDNA5 WMMA
// ---------------------------------------------------------------------------
typedef __attribute__((ext_vector_type(16))) __bf16 v16bf;
typedef __attribute__((ext_vector_type(8)))  __bf16 v8bf;
typedef __attribute__((ext_vector_type(8)))  float  v8f;

#define H 128
#define N_GENE 100000
#define N_VAR  200000
#define N_EDGE 500000
#define WPACK_ELEMS (H * H)   // 16384 bf16 per packed matrix

// Native bf16 convert (gfx1250: v_cvt_pk_bf16_f32)
static __device__ __forceinline__ __bf16 f2bf(float x) { return (__bf16)x; }

// ---------------------------------------------------------------------------
// Zero scratch (graph-capture-safe replacement for memset)
// ---------------------------------------------------------------------------
__global__ void zero_kernel(float* __restrict__ p, size_t n) {
    size_t i = (size_t)blockIdx.x * blockDim.x + threadIdx.x;
    size_t stride = (size_t)gridDim.x * blockDim.x;
    for (; i < n; i += stride) p[i] = 0.0f;
}

// ---------------------------------------------------------------------------
// Pack a 128x128 f32 weight matrix into B-fragment-ordered bf16:
//   dst[(((kc*8+oc)*32)+lane)*16 + e] = bf16( W[n*H + k] )
//   n = oc*16 + (lane&15),  k = kc*32 + 16*(lane>>4) + e
// ---------------------------------------------------------------------------
__global__ void pack_w_kernel(const float* __restrict__ W, __bf16* __restrict__ dst) {
    int idx = blockIdx.x * blockDim.x + threadIdx.x;
    if (idx >= WPACK_ELEMS) return;
    int e    = idx & 15;
    int lane = (idx >> 4) & 31;
    int oc   = (idx >> 9) & 7;
    int kc   = idx >> 12;
    int n = (oc << 4) + (lane & 15);
    int k = (kc << 5) + ((lane >> 4) << 4) + e;
    dst[idx] = f2bf(W[n * H + k]);
}

// ---------------------------------------------------------------------------
// Edge scatter-add: one wave (32 lanes) per edge; lane handles 4 floats.
// ---------------------------------------------------------------------------
__global__ void scatter_kernel(const float* __restrict__ x,
                               const int* __restrict__ src,
                               const int* __restrict__ dst,
                               float* __restrict__ agg,
                               float* __restrict__ cnt,
                               int nedges) {
    int wavesPerBlock = blockDim.x >> 5;
    int e = blockIdx.x * wavesPerBlock + (threadIdx.x >> 5);
    if (e >= nedges) return;
    int lane = threadIdx.x & 31;
    int s = src[e];
    int d = dst[e];
    const float4* srow = (const float4*)(x + (size_t)s * H);
    float4 v = srow[lane];
    float* drow = agg + (size_t)d * H + lane * 4;
    atomicAdd(drow + 0, v.x);
    atomicAdd(drow + 1, v.y);
    atomicAdd(drow + 2, v.z);
    atomicAdd(drow + 3, v.w);
    if (lane == 0) atomicAdd(cnt + d, 1.0f);
}

// ---------------------------------------------------------------------------
// Async memory->LDS copy of one 16B chunk (CDNA5 async path, ASYNCcnt)
// ---------------------------------------------------------------------------
static __device__ __forceinline__ void async_copy_b128(const void* gptr, void* lptr) {
    unsigned lds_off = (unsigned)(uintptr_t)lptr;       // LDS aperture: addr[31:0]
    uint64_t gaddr   = (uint64_t)(uintptr_t)gptr;
    asm volatile("global_load_async_to_lds_b128 %0, %1, off"
                 :: "v"(lds_off), "v"(gaddr) : "memory");
}
static __device__ __forceinline__ void async_wait0() {
    asm volatile("s_wait_asynccnt 0" ::: "memory");
}

// ---------------------------------------------------------------------------
// Load one pre-packed B fragment (32x16 bf16 tile of W^T) from LDS.
// f = kc*8 + oc (0..31)
// ---------------------------------------------------------------------------
static __device__ __forceinline__ v16bf load_bfrag(const __bf16* Wp, int f, int lane) {
    const v8bf* bp = (const v8bf*)(Wp + (((unsigned)f << 5) + lane) * 16);
    v8bf lo = bp[0];
    v8bf hi = bp[1];
    return __builtin_shufflevector(lo, hi,
        0, 1, 2, 3, 4, 5, 6, 7, 8, 9, 10, 11, 12, 13, 14, 15);
}

// ---------------------------------------------------------------------------
// One GEMM phase: acc += bf16(X[m,:]*scale) @ (packed W)^T over K=128.
// B fragments are software-pipelined: load f+1 before WMMA on f so the
// ds_load latency overlaps the matrix op.
// ---------------------------------------------------------------------------
static __device__ __forceinline__ void gemm_phase(v8f (&acc)[8],
                                                  const float* __restrict__ X,
                                                  float scale,
                                                  const __bf16* Wp,   // LDS
                                                  size_t m, int lane, int halfsel) {
    const float4* xrow = (const float4*)(X + m * H);
    v16bf bcur = load_bfrag(Wp, 0, lane);
#pragma unroll
    for (int kc = 0; kc < 4; ++kc) {
        int c0 = kc * 8 + halfsel * 4;          // float4 index of first run
        float4 x0 = xrow[c0];
        float4 x1 = xrow[c0 + 1];
        float4 x2 = xrow[c0 + 2];
        float4 x3 = xrow[c0 + 3];
        v16bf afrag;
        afrag[0]  = f2bf(x0.x * scale);
        afrag[1]  = f2bf(x0.y * scale);
        afrag[2]  = f2bf(x0.z * scale);
        afrag[3]  = f2bf(x0.w * scale);
        afrag[4]  = f2bf(x1.x * scale);
        afrag[5]  = f2bf(x1.y * scale);
        afrag[6]  = f2bf(x1.z * scale);
        afrag[7]  = f2bf(x1.w * scale);
        afrag[8]  = f2bf(x2.x * scale);
        afrag[9]  = f2bf(x2.y * scale);
        afrag[10] = f2bf(x2.z * scale);
        afrag[11] = f2bf(x2.w * scale);
        afrag[12] = f2bf(x3.x * scale);
        afrag[13] = f2bf(x3.y * scale);
        afrag[14] = f2bf(x3.z * scale);
        afrag[15] = f2bf(x3.w * scale);
#pragma unroll
        for (int oc = 0; oc < 8; ++oc) {
            int f = kc * 8 + oc;
            v16bf bnxt = (f < 31) ? load_bfrag(Wp, f + 1, lane) : bcur;
            acc[oc] = __builtin_amdgcn_wmma_f32_16x16x32_bf16(
                false, afrag, false, bcur, (short)0, acc[oc], false, false);
            bcur = bnxt;
        }
    }
}

// ---------------------------------------------------------------------------
// Fused SAGE GEMM:
//   out[n,:] = optReLU( (agg[n,:]/max(cnt[n],1)) @ Wl^T + xdst[n,:] @ Wr^T + b )
// agg == nullptr -> single-phase (adversarial head). N multiple of 16.
// Both packed weight matrices are staged in LDS (64 KB) via the CDNA5 async
// memory->LDS path and shared by all 8 waves of the block.
// ---------------------------------------------------------------------------
__global__ void sage_gemm_kernel(const float* __restrict__ agg,
                                 const float* __restrict__ cnt,
                                 const float* __restrict__ xdst,
                                 const __bf16* __restrict__ Wlp,
                                 const __bf16* __restrict__ Wrp,
                                 const float* __restrict__ bias,
                                 float* __restrict__ out,
                                 int N, int do_relu) {
    __shared__ __bf16 sW[2 * WPACK_ELEMS];   // 64 KB: [Wl packed | Wr packed]

    // Async cooperative fill (all threads participate BEFORE any early exit)
    {
        float4* sv = (float4*)sW;                       // 4096 float4 total
        if (Wlp) {
            const float4* g = (const float4*)Wlp;       // 2048 float4
            for (int i = threadIdx.x; i < 2048; i += blockDim.x)
                async_copy_b128(g + i, sv + i);
        }
        const float4* g = (const float4*)Wrp;
        for (int i = threadIdx.x; i < 2048; i += blockDim.x)
            async_copy_b128(g + i, sv + 2048 + i);
        async_wait0();
    }
    __syncthreads();

    int wavesPerBlock = blockDim.x >> 5;
    int tile = blockIdx.x * wavesPerBlock + (threadIdx.x >> 5);
    int ntiles = N >> 4;
    if (tile >= ntiles) return;              // wave-uniform: EXEC stays all-ones

    int lane    = threadIdx.x & 31;
    int halfsel = lane >> 4;
    int lm      = lane & 15;
    int row0    = tile << 4;
    size_t m    = (size_t)row0 + lm;         // row this lane loads for A

    v8f acc[8];
#pragma unroll
    for (int oc = 0; oc < 8; ++oc)
#pragma unroll
        for (int r = 0; r < 8; ++r) acc[oc][r] = 0.0f;

    if (agg) {
        float c = cnt[m];
        float inv = 1.0f / fmaxf(c, 1.0f);
        gemm_phase(acc, agg, inv, sW, m, lane, halfsel);
    }
    gemm_phase(acc, xdst, 1.0f, sW + WPACK_ELEMS, m, lane, halfsel);

    // Store D: lane holds col = oc*16 + lm, rows r + 8*halfsel
#pragma unroll
    for (int oc = 0; oc < 8; ++oc) {
        int col = (oc << 4) + lm;
        float bv = bias[col];
#pragma unroll
        for (int r = 0; r < 8; ++r) {
            int row = row0 + r + (halfsel << 3);
            float v = acc[oc][r] + bv;
            if (do_relu) v = fmaxf(v, 0.0f);
            out[(size_t)row * H + col] = v;
        }
    }
}

// ---------------------------------------------------------------------------
// Adversarial output: adv[n,o] = sum_k hdn[n,k]*A2[o,k] + b2[o]   (o < 8)
// ---------------------------------------------------------------------------
__global__ void adv_kernel(const float* __restrict__ hdn,
                           const float* __restrict__ A2,
                           const float* __restrict__ b2,
                           float* __restrict__ adv, int N) {
    int idx = blockIdx.x * blockDim.x + threadIdx.x;
    if (idx >= N * 8) return;
    int n = idx >> 3;
    int o = idx & 7;
    const float* hrow = hdn + (size_t)n * H;
    const float* arow = A2 + (size_t)o * H;
    float accv = b2[o];
#pragma unroll 8
    for (int k = 0; k < H; ++k) accv += hrow[k] * arow[k];
    adv[idx] = accv;
}

// ---------------------------------------------------------------------------
// Host-side launcher
// ---------------------------------------------------------------------------
extern "C" void kernel_launch(void* const* d_in, const int* in_sizes, int n_in,
                              void* d_out, int out_size, void* d_ws, size_t ws_size,
                              hipStream_t stream) {
    (void)in_sizes; (void)n_in; (void)out_size; (void)ws_size;

    const float* emb_gene = (const float*)d_in[0];
    const float* emb_var  = (const float*)d_in[1];
    const float* Wf[9] = {
        (const float*)d_in[2],   // Wl1_gv
        (const float*)d_in[4],   // Wr1_gv
        (const float*)d_in[5],   // Wl1_vg
        (const float*)d_in[7],   // Wr1_vg
        (const float*)d_in[8],   // Wl2_gv
        (const float*)d_in[10],  // Wr2_gv
        (const float*)d_in[11],  // Wl2_vg
        (const float*)d_in[13],  // Wr2_vg
        (const float*)d_in[14],  // A1
    };
    const float* bl1_gv = (const float*)d_in[3];
    const float* bl1_vg = (const float*)d_in[6];
    const float* bl2_gv = (const float*)d_in[9];
    const float* bl2_vg = (const float*)d_in[12];
    const float* b1 = (const float*)d_in[15];
    const float* A2 = (const float*)d_in[16];
    const float* b2 = (const float*)d_in[17];
    const int* src_gv = (const int*)d_in[18];
    const int* dst_gv = (const int*)d_in[19];
    const int* src_vg = (const int*)d_in[20];
    const int* dst_vg = (const int*)d_in[21];

    // Output layout: z_gene2 [100000*128] | z_var2 [200000*128] | adv [100000*8]
    float* zg  = (float*)d_out;                      // z_gene (L1), then z_gene2 in place
    float* zv  = (float*)d_out + (size_t)N_GENE * H; // z_var (L1), then z_var2 in place
    float* adv = (float*)d_out + (size_t)N_GENE * H + (size_t)N_VAR * H;

    // Workspace (floats):
    //   agg_var [N_VAR*H] | agg_gene [N_GENE*H] | cnt_var [N_VAR] | cnt_gene [N_GENE]
    //   | packed bf16 weights (9 * 16384 bf16)
    float* ws = (float*)d_ws;
    float* agg_var  = ws;
    float* agg_gene = ws + (size_t)N_VAR * H;
    float* cnt_var  = agg_gene + (size_t)N_GENE * H;
    float* cnt_gene = cnt_var + N_VAR;
    size_t zero_floats = (size_t)N_VAR * H + (size_t)N_GENE * H + N_VAR + N_GENE;
    __bf16* wpackBase = (__bf16*)(cnt_gene + N_GENE);
    __bf16* Wp[9];
    for (int i = 0; i < 9; ++i) Wp[i] = wpackBase + (size_t)i * WPACK_ELEMS;
    float* hdn = agg_gene;   // reused after layer-2 GEMMs consume agg_gene

    const int ZB = 256, ZG = 4096;
    const int SB = 256;                        // 8 waves/block, 1 edge/wave
    const int SG = (N_EDGE + 7) / 8;
    const int GB = 256;                        // 8 row-tiles per block
    const int gridVar  = (N_VAR / 16 + 7) / 8;
    const int gridGene = (N_GENE / 16 + 7) / 8;
    const int PG = (WPACK_ELEMS + 255) / 256;

    // Pack all WMMA weight matrices to fragment-ordered bf16 (tiny, stays in L2)
    for (int i = 0; i < 9; ++i)
        pack_w_kernel<<<PG, 256, 0, stream>>>(Wf[i], Wp[i]);

    // ----- Layer 1 -----
    zero_kernel<<<ZG, ZB, 0, stream>>>(ws, zero_floats);
    scatter_kernel<<<SG, SB, 0, stream>>>(emb_gene, src_gv, dst_gv, agg_var, cnt_var, N_EDGE);
    scatter_kernel<<<SG, SB, 0, stream>>>(emb_var,  src_vg, dst_vg, agg_gene, cnt_gene, N_EDGE);
    sage_gemm_kernel<<<gridVar,  GB, 0, stream>>>(agg_var,  cnt_var,  emb_var,  Wp[0], Wp[1], bl1_gv, zv, N_VAR, 1);
    sage_gemm_kernel<<<gridGene, GB, 0, stream>>>(agg_gene, cnt_gene, emb_gene, Wp[2], Wp[3], bl1_vg, zg, N_GENE, 1);

    // ----- Layer 2 -----
    zero_kernel<<<ZG, ZB, 0, stream>>>(ws, zero_floats);
    scatter_kernel<<<SG, SB, 0, stream>>>(zg, src_gv, dst_gv, agg_var, cnt_var, N_EDGE);
    scatter_kernel<<<SG, SB, 0, stream>>>(zv, src_vg, dst_vg, agg_gene, cnt_gene, N_EDGE);
    // in-place: each wave reads only its own 16 rows before writing them
    sage_gemm_kernel<<<gridVar,  GB, 0, stream>>>(agg_var,  cnt_var,  zv, Wp[4], Wp[5], bl2_gv, zv, N_VAR, 0);
    sage_gemm_kernel<<<gridGene, GB, 0, stream>>>(agg_gene, cnt_gene, zg, Wp[6], Wp[7], bl2_vg, zg, N_GENE, 0);

    // ----- Adversarial head -----
    sage_gemm_kernel<<<gridGene, GB, 0, stream>>>(nullptr, nullptr, zg, nullptr, Wp[8], b1, hdn, N_GENE, 1);
    adv_kernel<<<(N_GENE * 8 + 255) / 256, 256, 0, stream>>>(hdn, A2, b2, adv, N_GENE);
}
